// GANet_76304388981321
// MI455X (gfx1250) — matrix operand order repaired and checked by reading, not compiled, not run on hardware
//
#include <hip/hip_runtime.h>
#include <hip/hip_bf16.h>
#include <stdint.h>

// ---------------------------------------------------------------------------
// CDNA5 (gfx1250) GANet main path.
//  * dense layers: v_wmma_f32_16x16x32_bf16, 16x64 strip per wave (A reuse x4)
//  * packed weight panel staged to LDS once per block via TENSOR_LOAD_TO_LDS
//    (TDM, TENSORcnt) and served as B-fragments via ds_load_b128
//  * GAL edge softmax: per-edge VALU kernels + f32 global atomics
// ---------------------------------------------------------------------------

typedef __attribute__((ext_vector_type(16))) __bf16 v16bf;
typedef __attribute__((ext_vector_type(8)))  float  v8f;
typedef __attribute__((ext_vector_type(4)))  unsigned int uint32x4;
typedef __attribute__((ext_vector_type(8)))  int int32x8;
typedef __attribute__((ext_vector_type(4)))  int int32x4;

__device__ __forceinline__ unsigned short f2bf(float f) {
    unsigned int u = __float_as_uint(f);
    unsigned int r = (u + 0x7FFFu + ((u >> 16) & 1u)) >> 16;  // RNE
    return (unsigned short)r;
}

__device__ __forceinline__ void atomicMaxF(float* addr, float val) {
    int old = __float_as_int(*addr);
    while (__int_as_float(old) < val) {
        int assumed = old;
        old = atomicCAS((int*)addr, assumed, __float_as_int(val));
        if (old == assumed) break;
    }
}

// ---------------------------------------------------------------------------
// TDM stage: flat copy of `nbytes` (multiple of 8) from global -> LDS.
// D# built per CDNA5 ISA: 1-D tensor, data_size=8B,
// tensor_dim0 = tile_dim0 = nbytes/8, tensor_dim1 = tile_dim1 = 1, type=2.
// ---------------------------------------------------------------------------
__device__ __forceinline__ void tdm_stage_to_lds(const void* gptr, unsigned int lds_off,
                                                 unsigned int nbytes) {
    unsigned long long ga = (unsigned long long)(uintptr_t)gptr;
    unsigned int ne = nbytes >> 3;  // 8-byte elements (<= 65535 for our panels)

    uint32x4 g0;
    g0.x = 1u;                                         // count=1 (valid user D#)
    g0.y = lds_off;                                    // lds_addr
    g0.z = (unsigned int)ga;                           // global_addr[31:0]
    g0.w = (unsigned int)(ga >> 32) | (2u << 30);      // global_addr[56:32] | type=2

    int32x8 g1;
    g1[0] = (int)(3u << 16);                           // data_size=8B, mask=0
    g1[1] = (int)((ne & 0xFFFFu) << 16);               // tensor_dim0[15:0]
    g1[2] = (int)((ne >> 16) | (1u << 16));            // tensor_dim0[31:16] | tensor_dim1=1
    g1[3] = (int)((ne & 0xFFFFu) << 16);               // tile_dim0 (fits 16b: panels <=64KB)
    g1[4] = 1;                                         // tile_dim1=1, tile_dim2=0
    g1[5] = (int)ne;                                   // tensor_dim0_stride lo32
    g1[6] = 0;
    g1[7] = 0;

    int32x4 gz = {0, 0, 0, 0};
#if __clang_major__ >= 23
    int32x8 gz8 = {0, 0, 0, 0, 0, 0, 0, 0};
    __builtin_amdgcn_tensor_load_to_lds(g0, g1, gz, gz, gz8, 0);
#else
    __builtin_amdgcn_tensor_load_to_lds(g0, g1, gz, gz, 0);
#endif
    __builtin_amdgcn_s_wait_tensorcnt(0);
}

// ---------------------------------------------------------------------------
// Weight pre-pack: fp32 [Kin x M] row-major -> bf16 B-fragments.
// Tile (kt, nt): 32 lanes x 16 halves, lane L -> col nt*16+L%16,
// K = kt*32 + (L<16?0:8) + (j<8 ? j : j+8)  (16-bit B 32x16 layout, ISA 7.12.2).
// ---------------------------------------------------------------------------
__global__ void pack_w_kernel(const float* __restrict__ W,
                              unsigned short* __restrict__ out,
                              int Kin, int M, int Kpad, int Mpad) {
    int lane = threadIdx.x & 31;
    int tile = (blockIdx.x * blockDim.x + threadIdx.x) >> 5;
    int ntiles = Mpad >> 4, ktiles = Kpad >> 5;
    if (tile >= ntiles * ktiles) return;
    int kt  = tile / ntiles;
    int nt  = tile % ntiles;
    int col = nt * 16 + (lane & 15);
    int kb  = kt * 32 + ((lane < 16) ? 0 : 8);
    unsigned short* dst = out + (size_t)tile * 512 + lane * 16;
#pragma unroll
    for (int j = 0; j < 16; ++j) {
        int k = kb + (j < 8 ? j : j + 8);
        float w = (k < Kin && col < M) ? W[(size_t)k * M + col] : 0.0f;
        dst[j] = f2bf(w);
    }
}

// ---------------------------------------------------------------------------
// WMMA GEMM with LDS-staged weights.
//   grid: waves over (mtiles x ngroups); each wave -> 16 rows x (16*NT) cols.
//   Per k-step: 1 A-fragment load (2x b128), NT ds-load B fragments,
//   NT independent v_wmma_f32_16x16x32_bf16 (no RAW chain between them).
// ---------------------------------------------------------------------------
enum { GF_BIAS = 1, GF_RELU = 2, GF_RES = 4, GF_WBF = 8 };

template <int FLAGS, int NT, int LDX>
__global__ void gemm_wmma_kernel(const unsigned short* __restrict__ X,
                                 const unsigned short* __restrict__ Bp,
                                 const float* __restrict__ bias,
                                 const float* __restrict__ res,
                                 float* __restrict__ outF,
                                 unsigned short* __restrict__ outB, int ldob,
                                 int rows, int ncols) {
    extern __shared__ unsigned short smemB[];
    const int lane    = threadIdx.x & 31;
    const int wave    = threadIdx.x >> 5;
    const int nwaves  = blockDim.x >> 5;
    const int ntiles  = ncols >> 4;
    const int ktiles  = LDX >> 5;
    const int ngroups = ntiles / NT;
    const int mtiles  = rows >> 4;

    // ---- stage whole packed weight panel into LDS (one TDM op per block) ----
    if (wave == 0 && lane == 0) {
        tdm_stage_to_lds(Bp, (unsigned int)(uintptr_t)(void*)smemB,
                         (unsigned int)(ktiles * ntiles * 1024));
    }
    __syncthreads();

    int wtile = blockIdx.x * nwaves + wave;          // wave-uniform
    if (wtile >= mtiles * ngroups) return;           // EXEC all-ones below
    const int tm = wtile / ngroups;
    const int ng = wtile % ngroups;

    // A fragment addressing (16-bit A 16x32, ISA 7.12.2)
    const int row  = tm * 16 + (lane & 15);
    const int koff = (lane < 16) ? 0 : 8;
    const unsigned short* xrow = X + (size_t)row * LDX;

    v8f acc[NT];
#pragma unroll
    for (int t = 0; t < NT; ++t) acc[t] = (v8f){0.f,0.f,0.f,0.f,0.f,0.f,0.f,0.f};

#pragma unroll
    for (int kt = 0; kt < ktiles; ++kt) {
        union { uint4 q[2]; v16bf v; } ua;
        ua.q[0] = *(const uint4*)(xrow + kt * 32 + koff);
        ua.q[1] = *(const uint4*)(xrow + kt * 32 + 16 + koff);
        if (kt + 1 < ktiles)
            __builtin_prefetch(xrow + (kt + 1) * 32 + koff, 0, 1);
#pragma unroll
        for (int t = 0; t < NT; ++t) {
            union { uint4 q[2]; v16bf v; } ub;
            const unsigned short* bt =
                smemB + ((size_t)(kt * ntiles + ng * NT + t) * 512) + lane * 16;
            ub.q[0] = *(const uint4*)(bt);       // ds_load_b128
            ub.q[1] = *(const uint4*)(bt + 8);   // ds_load_b128
            acc[t] = __builtin_amdgcn_wmma_f32_16x16x32_bf16(
                false, ua.v, false, ub.v, (short)0, acc[t], false, false);
        }
    }

    // C/D layout: lane L -> N = L%16 ; VGPR r -> M = r + (L>=16 ? 8 : 0)
    const int r0 = (lane < 16) ? 0 : 8;
#pragma unroll
    for (int t = 0; t < NT; ++t) {
        const int col = (ng * NT + t) * 16 + (lane & 15);
        float bval = (FLAGS & GF_BIAS) ? bias[col] : 0.0f;
#pragma unroll
        for (int r = 0; r < 8; ++r) {
            int orow = tm * 16 + r0 + r;
            float v = acc[t][r] + bval;
            if (FLAGS & GF_RELU) v = fmaxf(v, 0.0f);
            if (FLAGS & GF_RES)  v += res[(size_t)orow * ncols + col];
            outF[(size_t)orow * ncols + col] = v;
            if (FLAGS & GF_WBF)  outB[(size_t)orow * ldob + col] = f2bf(v);
        }
    }
}

// ---------------------------------------------------------------------------
// Mesh-feature encoders + embed-input assembly.
// xin[N x 96] bf16 = [ colorFeat(32) | normalFeat(32) | pos(3) | zeros(29) ]
// ---------------------------------------------------------------------------
__global__ void encode_kernel(const float* __restrict__ pos,
                              const float* __restrict__ rgb,
                              const float* __restrict__ nrm,
                              const float* __restrict__ Wc, const float* __restrict__ bc,
                              const float* __restrict__ pec,
                              const float* __restrict__ Wn, const float* __restrict__ bn,
                              const float* __restrict__ pen,
                              unsigned short* __restrict__ xin, int n) {
    int id = blockIdx.x * blockDim.x + threadIdx.x;
    if (id >= n * 64) return;
    int node = id >> 6, c = id & 63;
    const float *W, *b, *pe, *src; int oc;
    if (c < 32) { W = Wc; b = bc; pe = pec; src = rgb; oc = c; }
    else        { W = Wn; b = bn; pe = pen; src = nrm; oc = c - 32; }
    float best = -3.0e38f;
#pragma unroll
    for (int t = 0; t < 4; ++t) {
        float acc = b[oc];
#pragma unroll
        for (int d = 0; d < 3; ++d) acc += src[node * 12 + t * 3 + d] * W[d * 32 + oc];
#pragma unroll
        for (int d = 0; d < 8; ++d) acc += pe[t * 8 + d] * W[(3 + d) * 32 + oc];
        best = fmaxf(best, fmaxf(acc, 0.0f));
    }
    xin[(size_t)node * 96 + c] = f2bf(best);
    if (c < 3) xin[(size_t)node * 96 + 64 + c] = f2bf(pos[node * 3 + c]);
}

// ---------------------------------------------------------------------------
// GAL edge kernels (C = 64). pe recomputed per pass (3xC FMA, L2-resident).
// ---------------------------------------------------------------------------
__device__ __forceinline__ float pe_val(const float* __restrict__ Wp,
                                        const float* __restrict__ bp,
                                        float d0, float d1, float d2, int cc) {
    return fmaxf(d0 * Wp[cc] + d1 * Wp[64 + cc] + d2 * Wp[128 + cc] + bp[cc], 0.0f);
}

__global__ void edge_logits_kernel(const int* __restrict__ src, const int* __restrict__ dst,
                                   const float* __restrict__ pos,
                                   const float* __restrict__ q, const float* __restrict__ k,
                                   const float* __restrict__ Wp, const float* __restrict__ bp,
                                   float* __restrict__ logits, float* __restrict__ m, int ne) {
    int e = blockIdx.x * blockDim.x + threadIdx.x;
    if (e >= ne) return;
    int s = src[e], d = dst[e];
    float d0 = pos[s * 3 + 0] - pos[d * 3 + 0];
    float d1 = pos[s * 3 + 1] - pos[d * 3 + 1];
    float d2 = pos[s * 3 + 2] - pos[d * 3 + 2];
    float acc = 0.0f;
    for (int cc = 0; cc < 64; ++cc) {
        float pe = pe_val(Wp, bp, d0, d1, d2, cc);
        acc += q[(size_t)d * 64 + cc] * (k[(size_t)s * 64 + cc] + pe);
    }
    acc *= 0.125f;   // 1/sqrt(64)
    logits[e] = acc;
    atomicMaxF(&m[d], acc);
}

__global__ void edge_expsum_kernel(const int* __restrict__ dst,
                                   float* __restrict__ logits,
                                   const float* __restrict__ m,
                                   float* __restrict__ den, int ne) {
    int e = blockIdx.x * blockDim.x + threadIdx.x;
    if (e >= ne) return;
    float w = __expf(logits[e] - m[dst[e]]);
    logits[e] = w;
    atomicAdd(&den[dst[e]], w);
}

__global__ void edge_aggregate_kernel(const int* __restrict__ src, const int* __restrict__ dst,
                                      const float* __restrict__ pos,
                                      const float* __restrict__ vmat,
                                      const float* __restrict__ Wp, const float* __restrict__ bp,
                                      const float* __restrict__ logits,
                                      const float* __restrict__ den,
                                      float* __restrict__ agg, int ne) {
    int e = blockIdx.x * blockDim.x + threadIdx.x;
    if (e >= ne) return;
    int s = src[e], d = dst[e];
    float a = logits[e] / (den[d] + 1e-12f);
    float d0 = pos[s * 3 + 0] - pos[d * 3 + 0];
    float d1 = pos[s * 3 + 1] - pos[d * 3 + 1];
    float d2 = pos[s * 3 + 2] - pos[d * 3 + 2];
    for (int cc = 0; cc < 64; ++cc) {
        float pe = pe_val(Wp, bp, d0, d1, d2, cc);
        atomicAdd(&agg[(size_t)d * 64 + cc], a * (vmat[(size_t)s * 64 + cc] + pe));
    }
}

// ---------------------------------------------------------------------------
// Utility kernels
// ---------------------------------------------------------------------------
__global__ void fill_f32_kernel(float* __restrict__ p, float v, int n) {
    int i = blockIdx.x * blockDim.x + threadIdx.x;
    if (i < n) p[i] = v;
}

__global__ void f2bf_kernel(const float* __restrict__ in, unsigned short* __restrict__ out,
                            int rows, int C, int Cpad) {
    int i = blockIdx.x * blockDim.x + threadIdx.x;
    if (i >= rows * Cpad) return;
    int r = i / Cpad, c = i % Cpad;
    out[i] = (c < C) ? f2bf(in[(size_t)r * C + c]) : (unsigned short)0;
}

// ---------------------------------------------------------------------------
// Input index map (jax tree-flatten: top dict in insertion order,
// nested dicts in sorted-key order; capitals sort before lowercase).
// ---------------------------------------------------------------------------
#define IDX_POS      0
#define IDX_RGB      1
#define IDX_NRM      2
#define IDX_EDGE     3
#define IDX_C_W      4
#define IDX_C_B      5
#define IDX_C_PE     6
// convs 7..54, dec 55..66
#define IDX_EMB_W    67
#define IDX_EMB_B    68
#define IDX_H0_W     69
#define IDX_H0_B     70
#define IDX_H1_W     71
#define IDX_H1_B     72
#define IDX_H2_W     73
#define IDX_H2_B     74
#define IDX_LG_WK    75
#define IDX_LG_WO    76
#define IDX_LG_WP    77
#define IDX_LG_WQ    78
#define IDX_LG_WV    79
#define IDX_LG_BP    80
#define IDX_N_W      81
#define IDX_N_B      82
#define IDX_N_PE     83
#define IDX_R0_W     84
#define IDX_R0_B     85
#define IDX_R1_W     86
#define IDX_R1_B     87

extern "C" void kernel_launch(void* const* d_in, const int* in_sizes, int n_in,
                              void* d_out, int out_size, void* d_ws, size_t ws_size,
                              hipStream_t stream) {
    const int N  = in_sizes[IDX_POS] / 3;       // 20000
    const int NE = in_sizes[IDX_EDGE] / 2;      // 320000

    const float* pos = (const float*)d_in[IDX_POS];
    const float* rgb = (const float*)d_in[IDX_RGB];
    const float* nrm = (const float*)d_in[IDX_NRM];
    const int*   esrc = (const int*)d_in[IDX_EDGE];
    const int*   edst = esrc + NE;

    // ---- workspace carve-up (256B-aligned bump allocator) ----
    char* base = (char*)d_ws;
    size_t off = 0;
    auto alloc = [&](size_t bytes) -> char* {
        off = (off + 255) & ~(size_t)255;
        char* p = base + off;
        off += bytes;
        return p;
    };
    unsigned short* xin    = (unsigned short*)alloc((size_t)N * 96 * 2);
    float*          xf     = (float*)         alloc((size_t)N * 64 * 4);
    unsigned short* xbf    = (unsigned short*)alloc((size_t)N * 64 * 2);
    float*          qf     = (float*)         alloc((size_t)N * 64 * 4);
    float*          kf     = (float*)         alloc((size_t)N * 64 * 4);
    float*          vf     = (float*)         alloc((size_t)N * 64 * 4);
    float*          agg    = (float*)         alloc((size_t)N * 64 * 4);
    unsigned short* aggbf  = (unsigned short*)alloc((size_t)N * 64 * 2);
    float*          x2f    = (float*)         alloc((size_t)N * 64 * 4);
    unsigned short* x2bf   = (unsigned short*)alloc((size_t)N * 64 * 2);
    float*          x3f    = (float*)         alloc((size_t)N * 64 * 4);
    unsigned short* x3bf   = (unsigned short*)alloc((size_t)N * 64 * 2);
    float*          x4f    = (float*)         alloc((size_t)N * 64 * 4);
    unsigned short* x4bf   = (unsigned short*)alloc((size_t)N * 64 * 2);
    float*          h0f    = (float*)         alloc((size_t)N * 128 * 4);
    unsigned short* h0bf   = (unsigned short*)alloc((size_t)N * 128 * 2);
    float*          h1f    = (float*)         alloc((size_t)N * 128 * 4);
    unsigned short* h1bf   = (unsigned short*)alloc((size_t)N * 128 * 2);
    float*          logits = (float*)         alloc((size_t)NE * 4);
    float*          mmax   = (float*)         alloc((size_t)N * 4);
    float*          den    = (float*)         alloc((size_t)N * 4);
    unsigned short* pWe    = (unsigned short*)alloc(96 * 64 * 2);
    unsigned short* pWq    = (unsigned short*)alloc(64 * 64 * 2);
    unsigned short* pWk    = (unsigned short*)alloc(64 * 64 * 2);
    unsigned short* pWv    = (unsigned short*)alloc(64 * 64 * 2);
    unsigned short* pWo    = (unsigned short*)alloc(64 * 64 * 2);
    unsigned short* pWr0   = (unsigned short*)alloc(64 * 64 * 2);
    unsigned short* pWr1   = (unsigned short*)alloc(64 * 64 * 2);
    unsigned short* pWh0   = (unsigned short*)alloc(64 * 128 * 2);
    unsigned short* pWh1   = (unsigned short*)alloc(128 * 128 * 2);
    unsigned short* pWh2   = (unsigned short*)alloc(128 * 16 * 2);
    (void)ws_size;

    // ---- weight packing ----
    auto pack = [&](const void* W, unsigned short* out, int Kin, int M, int Kp, int Mp) {
        int tiles = (Kp / 32) * (Mp / 16);
        pack_w_kernel<<<(tiles * 32 + 255) / 256, 256, 0, stream>>>(
            (const float*)W, out, Kin, M, Kp, Mp);
    };
    pack(d_in[IDX_EMB_W], pWe, 67, 64, 96, 64);
    pack(d_in[IDX_LG_WQ], pWq, 64, 64, 64, 64);
    pack(d_in[IDX_LG_WK], pWk, 64, 64, 64, 64);
    pack(d_in[IDX_LG_WV], pWv, 64, 64, 64, 64);
    pack(d_in[IDX_LG_WO], pWo, 64, 64, 64, 64);
    pack(d_in[IDX_R0_W],  pWr0, 64, 64, 64, 64);
    pack(d_in[IDX_R1_W],  pWr1, 64, 64, 64, 64);
    pack(d_in[IDX_H0_W],  pWh0, 64, 128, 64, 128);
    pack(d_in[IDX_H1_W],  pWh1, 128, 128, 128, 128);
    pack(d_in[IDX_H2_W],  pWh2, 128, 16, 128, 16);

    // ---- encoders -> xin ----
    hipMemsetAsync(xin, 0, (size_t)N * 96 * 2, stream);
    encode_kernel<<<(N * 64 + 255) / 256, 256, 0, stream>>>(
        pos, rgb, nrm,
        (const float*)d_in[IDX_C_W], (const float*)d_in[IDX_C_B], (const float*)d_in[IDX_C_PE],
        (const float*)d_in[IDX_N_W], (const float*)d_in[IDX_N_B], (const float*)d_in[IDX_N_PE],
        xin, N);

    const int mtiles = N / 16;
    auto blocks_for = [&](int ngroups) { return (mtiles * ngroups + 3) / 4; };
    auto shb = [](int ktiles, int ntiles) { return (size_t)ktiles * ntiles * 1024; };

    // ---- embed: x = relu(xin @ We + b) ----  ldx=96, ncols=64, NT=4
    gemm_wmma_kernel<GF_BIAS | GF_RELU | GF_WBF, 4, 96>
        <<<blocks_for(1), 128, shb(3, 4), stream>>>(
            xin, pWe, (const float*)d_in[IDX_EMB_B], nullptr, xf, xbf, 64, N, 64);

    // ---- local GAL: q/k/v projections ----  ldx=64, ncols=64
    gemm_wmma_kernel<0, 4, 64><<<blocks_for(1), 128, shb(2, 4), stream>>>(
        xbf, pWq, nullptr, nullptr, qf, nullptr, 64, N, 64);
    gemm_wmma_kernel<0, 4, 64><<<blocks_for(1), 128, shb(2, 4), stream>>>(
        xbf, pWk, nullptr, nullptr, kf, nullptr, 64, N, 64);
    gemm_wmma_kernel<0, 4, 64><<<blocks_for(1), 128, shb(2, 4), stream>>>(
        xbf, pWv, nullptr, nullptr, vf, nullptr, 64, N, 64);

    // ---- segment softmax + aggregation over edges ----
    fill_f32_kernel<<<(N + 255) / 256, 256, 0, stream>>>(mmax, -3.0e38f, N);
    hipMemsetAsync(den, 0, (size_t)N * 4, stream);
    hipMemsetAsync(agg, 0, (size_t)N * 64 * 4, stream);
    const float* Wp = (const float*)d_in[IDX_LG_WP];
    const float* bp = (const float*)d_in[IDX_LG_BP];
    edge_logits_kernel<<<(NE + 255) / 256, 256, 0, stream>>>(
        esrc, edst, pos, qf, kf, Wp, bp, logits, mmax, NE);
    edge_expsum_kernel<<<(NE + 255) / 256, 256, 0, stream>>>(edst, logits, mmax, den, NE);
    edge_aggregate_kernel<<<(NE + 255) / 256, 256, 0, stream>>>(
        esrc, edst, pos, vf, Wp, bp, logits, den, agg, NE);

    // ---- x2 = x + relu(agg @ Wo) ----
    f2bf_kernel<<<(N * 64 + 255) / 256, 256, 0, stream>>>(agg, aggbf, N, 64, 64);
    gemm_wmma_kernel<GF_RELU | GF_RES | GF_WBF, 4, 64>
        <<<blocks_for(1), 128, shb(2, 4), stream>>>(
            aggbf, pWo, nullptr, xf, x2f, x2bf, 64, N, 64);

    // ---- res MLPs: x = x + relu(x @ W + b) ----
    gemm_wmma_kernel<GF_BIAS | GF_RELU | GF_RES | GF_WBF, 4, 64>
        <<<blocks_for(1), 128, shb(2, 4), stream>>>(
            x2bf, pWr0, (const float*)d_in[IDX_R0_B], x2f, x3f, x3bf, 64, N, 64);
    gemm_wmma_kernel<GF_BIAS | GF_RELU | GF_RES | GF_WBF, 4, 64>
        <<<blocks_for(1), 128, shb(2, 4), stream>>>(
            x3bf, pWr1, (const float*)d_in[IDX_R1_B], x3f, x4f, x4bf, 64, N, 64);

    // ---- head: 64 -> 128 -> 128 -> 16 ----
    gemm_wmma_kernel<GF_BIAS | GF_RELU | GF_WBF, 4, 64>
        <<<blocks_for(2), 128, shb(2, 8), stream>>>(
            x4bf, pWh0, (const float*)d_in[IDX_H0_B], nullptr, h0f, h0bf, 128, N, 128);
    gemm_wmma_kernel<GF_BIAS | GF_RELU | GF_WBF, 4, 128>
        <<<blocks_for(2), 128, shb(4, 8), stream>>>(
            h0bf, pWh1, (const float*)d_in[IDX_H1_B], nullptr, h1f, h1bf, 128, N, 128);
    gemm_wmma_kernel<GF_BIAS, 1, 128>
        <<<blocks_for(1), 128, shb(4, 1), stream>>>(
            h1bf, pWh2, (const float*)d_in[IDX_H2_B], nullptr, (float*)d_out, nullptr, 16, N, 16);

    (void)n_in; (void)out_size;
}
